// Net_69363721830867
// MI455X (gfx1250) — matrix operand order repaired and checked by reading, not compiled, not run on hardware
//
#include <hip/hip_runtime.h>

// ---------------------------------------------------------------------------
// GCN pipeline for MI455X (gfx1250, wave32).
// Dense GEMMs run on v_wmma_f32_16x16x32_f16 (f16 in, f32 accumulate).
// Scatter phase uses f32 global atomics (served by L2 atomic units; h fits L2).
// Workspace requirement: ~471 MB.
// ---------------------------------------------------------------------------

#define NNODES   100000
#define KDIM     288        // 256 (x) + 32 (static), multiple of 32
#define FDIM     256
#define NGR      50

typedef __attribute__((ext_vector_type(16))) _Float16 v16h;
typedef __attribute__((ext_vector_type(8)))  _Float16 v8h;
typedef __attribute__((ext_vector_type(8)))  float    v8f;

// ------------------------------ utility kernels ----------------------------

__global__ void fill_f32(float* __restrict__ p, float v, long long n) {
  long long i = (long long)blockIdx.x * blockDim.x + threadIdx.x;
  if (i < n) p[i] = v;
}

__global__ void count_deg(const int* __restrict__ dst, int E, float* __restrict__ deg) {
  int e = blockIdx.x * blockDim.x + threadIdx.x;
  if (e < E) atomicAdd(&deg[dst[e]], 1.0f);
}

__global__ void rsqrt_k(float* __restrict__ p, long long n) {
  long long i = (long long)blockIdx.x * blockDim.x + threadIdx.x;
  if (i < n) p[i] = rsqrtf(p[i]);
}

// Pack W [Kpad x 256] (f32, rows >= Ksrc are zero) into WMMA B-fragment order:
// flat = (((k32*16 + ntile)*32 + lane)*16 + j)
//   lane<16 : column n=lane,    K = k32*32 + {0..7,16..23}[j]
//   lane>=16: column n=lane-16, K = k32*32 + {8..15,24..31}[j]
__global__ void pack_w(const float* __restrict__ W, _Float16* __restrict__ Bp,
                       int Kpad, int Ksrc) {
  int flat = blockIdx.x * blockDim.x + threadIdx.x;
  int total = Kpad * 256;
  if (flat >= total) return;
  int j    = flat & 15;
  int lane = (flat >> 4) & 31;
  int nt   = (flat >> 9) & 15;
  int k32  = flat >> 13;
  int n    = lane & 15;
  int kst  = (lane >> 4) * 8;
  int krow = k32 * 32 + kst + (j & 7) + ((j >> 3) << 4);
  int ncol = nt * 16 + n;
  float v = (krow < Ksrc) ? W[(long long)krow * 256 + ncol] : 0.0f;
  Bp[flat] = (_Float16)v;
}

// static feature columns 256..287 of A (written once per launch)
__global__ void build_static(const float* __restrict__ fs, _Float16* __restrict__ a16) {
  long long i = (long long)blockIdx.x * blockDim.x + threadIdx.x;
  if (i >= (long long)NNODES * 32) return;
  long long node = i >> 5;
  int c = (int)(i & 31);
  a16[node * KDIM + 256 + c] = (_Float16)fs[i];
}

// initial conv A: cols 0..15 = x_in, cols 16..31 = 0  (K=32, lda=KDIM)
__global__ void build_a_init(const float* __restrict__ xin, _Float16* __restrict__ a16) {
  long long i = (long long)blockIdx.x * blockDim.x + threadIdx.x;
  if (i >= (long long)NNODES * 32) return;
  long long node = i >> 5;
  int c = (int)(i & 31);
  float v = (c < 16) ? xin[node * 16 + c] : 0.0f;
  a16[node * KDIM + c] = (_Float16)v;
}

// main conv A: cols 0..255 = current x (f32 -> f16)
__global__ void build_a_main(const float* __restrict__ x, _Float16* __restrict__ a16) {
  long long i = (long long)blockIdx.x * blockDim.x + threadIdx.x;   // over N*64
  if (i >= (long long)NNODES * 64) return;
  long long node = i >> 6;
  int c4 = (int)(i & 63);
  float4 v = *(const float4*)(x + node * 256 + c4 * 4);
  _Float16* o = a16 + node * KDIM + c4 * 4;
  o[0] = (_Float16)v.x; o[1] = (_Float16)v.y;
  o[2] = (_Float16)v.z; o[3] = (_Float16)v.w;
}

// ------------------------------ WMMA GEMM ----------------------------------
// C[M,256] = A[M,K] @ W[K,256]   (bias added later by init_acc)
// Block = 256 threads = 8 waves: 2 m-subtiles x 4 n-strips (64 cols each).
// Grid.x = M/32.
__global__ __launch_bounds__(256)
void gemm16(const _Float16* __restrict__ A, const _Float16* __restrict__ Bp,
            float* __restrict__ C, int K, int lda) {
  int lane   = threadIdx.x & 31;
  int wave   = threadIdx.x >> 5;
  int msub   = wave >> 2;                 // 0..1
  int nstrip = wave & 3;                  // 0..3
  int mtile  = blockIdx.x * 2 + msub;     // 16-row tile index
  int mrow   = mtile * 16 + (lane & 15);
  int khalf  = (lane >> 4) * 8;           // A layout: lanes 0-15 hold K{0..7,16..23}
  const _Float16* arow = A + (long long)mrow * lda;

  v8f acc0 = {}, acc1 = {}, acc2 = {}, acc3 = {};

  for (int kb = 0; kb < K; kb += 32) {
    v8h lo = *(const v8h*)(arow + kb + khalf);
    v8h hi = *(const v8h*)(arow + kb + 16 + khalf);
    v16h af;
#pragma unroll
    for (int t = 0; t < 8; ++t) { af[t] = lo[t]; af[t + 8] = hi[t]; }

    long long bbase = (((long long)(kb >> 5) * 16 + nstrip * 4) * 32 + lane) * 16;
    v16h b0 = *(const v16h*)(Bp + bbase);
    v16h b1 = *(const v16h*)(Bp + bbase + 512);
    v16h b2 = *(const v16h*)(Bp + bbase + 1024);
    v16h b3 = *(const v16h*)(Bp + bbase + 1536);

    acc0 = __builtin_amdgcn_wmma_f32_16x16x32_f16(false, af, false, b0, (short)0, acc0, false, false);
    acc1 = __builtin_amdgcn_wmma_f32_16x16x32_f16(false, af, false, b1, (short)0, acc1, false, false);
    acc2 = __builtin_amdgcn_wmma_f32_16x16x32_f16(false, af, false, b2, (short)0, acc2, false, false);
    acc3 = __builtin_amdgcn_wmma_f32_16x16x32_f16(false, af, false, b3, (short)0, acc3, false, false);
  }

  // C/D layout: VGPR r -> row (lane<16 ? r : r+8), col = lane&15
  int outrow = mtile * 16 + (lane >> 4) * 8;
  int outcol = nstrip * 64 + (lane & 15);
  float* cp = C + (long long)outrow * 256 + outcol;
#pragma unroll
  for (int r = 0; r < 8; ++r) {
    cp[(long long)r * 256 +  0] = acc0[r];
    cp[(long long)r * 256 + 16] = acc1[r];
    cp[(long long)r * 256 + 32] = acc2[r];
    cp[(long long)r * 256 + 48] = acc3[r];
  }
}

// ------------------------------ conv epilogue ------------------------------

// out[i,:] = dinv[i]^2 * h[i,:] + bias  (self-loop term + bias; scatter adds edges)
__global__ void init_acc(const float* __restrict__ h, const float* __restrict__ dinv,
                         const float* __restrict__ bias, float* __restrict__ out) {
  long long i = (long long)blockIdx.x * blockDim.x + threadIdx.x;   // N*64
  if (i >= (long long)NNODES * 64) return;
  long long node = i >> 6;
  int c4 = (int)(i & 63);
  float w = dinv[node]; w = w * w;
  float4 hv = *(const float4*)(h + node * 256 + c4 * 4);
  float4 bv = *(const float4*)(bias + c4 * 4);
  float4 o;
  o.x = w * hv.x + bv.x; o.y = w * hv.y + bv.y;
  o.z = w * hv.z + bv.z; o.w = w * hv.w + bv.w;
  *(float4*)(out + node * 256 + c4 * 4) = o;
}

// acc[dst,:] += dinv[src]*dinv[dst] * h[src,:]   (64 lanes per edge, float4 each)
__global__ void scatter_add(const int* __restrict__ src, const int* __restrict__ dst,
                            const float* __restrict__ dinv, const float* __restrict__ h,
                            float* __restrict__ acc, int E) {
  int e = blockIdx.x * 4 + (threadIdx.x >> 6);
  if (e >= E) return;
  int t = threadIdx.x & 63;
  int s = src[e], d = dst[e];
  float w = dinv[s] * dinv[d];
  float4 v = *(const float4*)(h + (long long)s * 256 + t * 4);
  float* ap = acc + (long long)d * 256 + t * 4;
  atomicAdd(ap + 0, w * v.x);
  atomicAdd(ap + 1, w * v.y);
  atomicAdd(ap + 2, w * v.z);
  atomicAdd(ap + 3, w * v.w);
}

// x[i,:] = srcbuf[i,:] where layers[i]==target
__global__ void mask_assign(float* __restrict__ x, const float* __restrict__ srcbuf,
                            const int* __restrict__ layers, int target) {
  long long i = (long long)blockIdx.x * blockDim.x + threadIdx.x;   // N*64
  if (i >= (long long)NNODES * 64) return;
  long long node = i >> 6;
  int c4 = (int)(i & 63);
  if (layers[node] == target) {
    *(float4*)(x + node * 256 + c4 * 4) =
        *(const float4*)(srcbuf + node * 256 + c4 * 4);
  }
}

__global__ void relu_k(float* __restrict__ x) {
  long long i = (long long)blockIdx.x * blockDim.x + threadIdx.x;   // N*64
  if (i >= (long long)NNODES * 64) return;
  float4 v = *(const float4*)(x + i * 4);
  v.x = fmaxf(v.x, 0.f); v.y = fmaxf(v.y, 0.f);
  v.z = fmaxf(v.z, 0.f); v.w = fmaxf(v.w, 0.f);
  *(float4*)(x + i * 4) = v;
}

__global__ void pool_k(const float* __restrict__ x, const int* __restrict__ batch,
                       float* __restrict__ pooled) {
  long long i = (long long)blockIdx.x * blockDim.x + threadIdx.x;   // N*64
  if (i >= (long long)NNODES * 64) return;
  long long node = i >> 6;
  int c4 = (int)(i & 63);
  int g = batch[node];
  float4 v = *(const float4*)(x + node * 256 + c4 * 4);
  float* pp = pooled + (long long)g * 256 + c4 * 4;
  atomicAdd(pp + 0, v.x); atomicAdd(pp + 1, v.y);
  atomicAdd(pp + 2, v.z); atomicAdd(pp + 3, v.w);
}

__global__ void final_k(const float* __restrict__ pooled, const float* __restrict__ Wl,
                        const float* __restrict__ bl, float* __restrict__ out) {
  __shared__ float red[256];
  int g = blockIdx.x, t = threadIdx.x;
  red[t] = pooled[(long long)g * 256 + t] * Wl[t];
  __syncthreads();
  for (int s = 128; s > 0; s >>= 1) {
    if (t < s) red[t] += red[t + s];
    __syncthreads();
  }
  if (t == 0) out[g] = red[0] + bl[0];
}

// ------------------------------ host orchestration -------------------------

extern "C" void kernel_launch(void* const* d_in, const int* in_sizes, int n_in,
                              void* d_out, int out_size, void* d_ws, size_t ws_size,
                              hipStream_t stream) {
  const float* x_in  = (const float*)d_in[0];
  const float* fstat = (const float*)d_in[1];
  const int*   eidx  = (const int*)d_in[2];
  const int*   inner = (const int*)d_in[3];
  const int*   fwd   = (const int*)d_in[4];
  // d_in[5] backward_edges: only feeds dead compute in the reference -> unused
  const int*   layers= (const int*)d_in[6];
  const int*   batch = (const int*)d_in[7];
  const float* W_up  = (const float*)d_in[8];
  const float* b_up  = (const float*)d_in[9];
  const float* W_in  = (const float*)d_in[10];
  const float* b_in  = (const float*)d_in[11];
  const float* W_fw  = (const float*)d_in[12];
  const float* b_fw  = (const float*)d_in[13];
  const float* W_l   = (const float*)d_in[14];
  const float* b_l   = (const float*)d_in[15];
  float* out = (float*)d_out;

  // workspace carve (~471 MB)
  char* p = (char*)d_ws;
  auto carve = [&](size_t bytes) -> void* {
    void* r = (void*)p; p += (bytes + 255) & ~(size_t)255; return r;
  };
  float*    dinv   = (float*)   carve((size_t)8 * NNODES * 4);
  _Float16* a16    = (_Float16*)carve((size_t)NNODES * KDIM * 2);
  float*    h      = (float*)   carve((size_t)NNODES * 256 * 4);
  float*    acc    = (float*)   carve((size_t)NNODES * 256 * 4);
  float*    pinner = (float*)   carve((size_t)NNODES * 256 * 4);
  float*    x      = (float*)   carve((size_t)NNODES * 256 * 4);
  _Float16* WupP   = (_Float16*)carve((size_t)32  * 256 * 2);
  _Float16* WinP   = (_Float16*)carve((size_t)288 * 256 * 2);
  _Float16* WfwP   = (_Float16*)carve((size_t)288 * 256 * 2);
  float*    pooled = (float*)   carve((size_t)NGR * 256 * 4);

  const int G64   = (NNODES * 64) / 256;   // 25000 blocks over N*64 lanes
  const int G32   = (NNODES * 32) / 256;   // 12500 blocks over N*32 lanes
  const int GGEMM = NNODES / 32;           // 3125

  // ---- degrees: deg=1 (self loop) + in-degree; then rsqrt.  8 live lists. ----
  long long n8 = (long long)8 * NNODES;
  fill_f32<<<(int)((n8 + 255) / 256), 256, 0, stream>>>(dinv, 1.0f, n8);
  const int* dlist[8]; int dE[8];
  dlist[0] = eidx + 400000;                 dE[0] = 400000;            // edge_index
  for (int il = 0; il < 4; ++il) { dlist[1 + il] = inner + il * 600000 + 300000; dE[1 + il] = 300000; }
  for (int il = 0; il < 3; ++il) { dlist[5 + il] = fwd   + il * 600000 + 300000; dE[5 + il] = 300000; }
  for (int j = 0; j < 8; ++j)
    count_deg<<<(dE[j] + 255) / 256, 256, 0, stream>>>(dlist[j], dE[j], dinv + (long long)j * NNODES);
  rsqrt_k<<<(int)((n8 + 255) / 256), 256, 0, stream>>>(dinv, n8);

  // ---- pack weights into WMMA B-fragment order; static A columns ----
  pack_w<<<(32  * 256 + 255) / 256, 256, 0, stream>>>(W_up, WupP, 32, 16);
  pack_w<<<(288 * 256 + 255) / 256, 256, 0, stream>>>(W_in, WinP, 288, 288);
  pack_w<<<(288 * 256 + 255) / 256, 256, 0, stream>>>(W_fw, WfwP, 288, 288);
  build_static<<<G32, 256, 0, stream>>>(fstat, a16);

  // ---- initial conv: x = gcn(x_in, edge_index, W_up, b_up) ----
  build_a_init<<<G32, 256, 0, stream>>>(x_in, a16);
  gemm16<<<GGEMM, 256, 0, stream>>>(a16, WupP, h, 32, KDIM);
  init_acc<<<G64, 256, 0, stream>>>(h, dinv, b_up, x);
  scatter_add<<<(400000 + 3) / 4, 256, 0, stream>>>(eidx, eidx + 400000, dinv, h, x, 400000);

  // one full 288->256 conv into `target`
  auto conv = [&](const _Float16* Wp, const float* bias, const int* srcp, const int* dstp,
                  int E, const float* dv, float* target) {
    build_a_main<<<G64, 256, 0, stream>>>(x, a16);
    gemm16<<<GGEMM, 256, 0, stream>>>(a16, Wp, h, 288, KDIM);
    init_acc<<<G64, 256, 0, stream>>>(h, dv, bias, target);
    scatter_add<<<(E + 3) / 4, 256, 0, stream>>>(srcp, dstp, dv, h, target, E);
  };

  for (int prop = 0; prop < 2; ++prop) {
    // forward sweep
    for (int il = 0; il < 4; ++il) {
      float* tgt = (il == 3) ? pinner : acc;   // il==3 inner conv is the "stale partial_inner"
      conv(WinP, b_in, inner + il * 600000, inner + il * 600000 + 300000,
           300000, dinv + (long long)(1 + il) * NNODES, tgt);
      mask_assign<<<G64, 256, 0, stream>>>(x, tgt, layers, il);
      if (il < 3) {   // il==3 forward conv masked by layers==4 (never true) -> dead
        conv(WfwP, b_fw, fwd + il * 600000, fwd + il * 600000 + 300000,
             300000, dinv + (long long)(5 + il) * NNODES, acc);
        mask_assign<<<G64, 256, 0, stream>>>(x, acc, layers, il + 1);
      }
    }
    relu_k<<<G64, 256, 0, stream>>>(x);
    // backward sweep (uses stale pinner; conv on backward_edges is dead in ref)
    for (int il = 3; il >= 0; --il) {
      if (il > 0)
        mask_assign<<<G64, 256, 0, stream>>>(x, pinner, layers, il - 1);
      conv(WinP, b_in, inner + il * 600000, inner + il * 600000 + 300000,
           300000, dinv + (long long)(1 + il) * NNODES, acc);
      mask_assign<<<G64, 256, 0, stream>>>(x, acc, layers, il);
    }
    relu_k<<<G64, 256, 0, stream>>>(x);
  }

  // ---- pool by graph, then [G,256] @ [256,1] + b ----
  fill_f32<<<(NGR * 256 + 255) / 256, 256, 0, stream>>>(pooled, 0.0f, NGR * 256);
  pool_k<<<G64, 256, 0, stream>>>(x, batch, pooled);
  final_k<<<NGR, 256, 0, stream>>>(pooled, W_l, b_l, out);
}
